// PointTransformerCls_72919954752185
// MI455X (gfx1250) — matrix operand order repaired and checked by reading, not compiled, not run on hardware
//
#include <hip/hip_runtime.h>
#include <math.h>

// ---------------------------------------------------------------------------
// Point Transformer classifier forward for MI455X (gfx1250, wave32, WMMA).
// GEMM-heavy stages use V_WMMA_F32_16X16X32_F16 (f16 in, f32 accumulate) with
// async global->LDS staging of the A panel; irregular stages (FPS / kNN /
// softmax / BN / gather / maxpool) are VALU+LDS kernels.
// ---------------------------------------------------------------------------

typedef __attribute__((ext_vector_type(16))) _Float16 v16h;
typedef __attribute__((ext_vector_type(8)))  float    v8f;

// types for the async global->LDS builtin (param types per hipcc diagnostic)
typedef int v4i __attribute__((vector_size(16)));
typedef __attribute__((address_space(1))) v4i* gv4i_p;   // global (AS1)
typedef __attribute__((address_space(3))) v4i* lv4i_p;   // LDS (AS3)

#define TPB 256

#if defined(__HIP_DEVICE_COMPILE__) && __has_builtin(__builtin_amdgcn_global_load_async_to_lds_b128)
#define USE_ASYNC_LDS 1
#else
#define USE_ASYNC_LDS 0
#endif

__device__ __forceinline__ void wait_async_lds() {
#if USE_ASYNC_LDS
#if __has_builtin(__builtin_amdgcn_s_wait_asynccnt)
    __builtin_amdgcn_s_wait_asynccnt(0);
#else
    asm volatile("s_wait_asynccnt 0x0" ::: "memory");
#endif
#endif
}

// ============================ WMMA GEMM ====================================
// Y[M,Nc] = act( X[M,Kc] @ W[Kc,Nc] + bias (+ res) ), all f32 in memory.
// Block = 64 threads = 2 waves; each wave owns one 16x16 D tile; the 16x32
// A panel is staged in LDS (async global->LDS B128 when available).
// Requires M%16==0, Kc%32==0, Nc%32==0 (true for every call site below).
__global__ __launch_bounds__(64) void k_wmma_gemm(
    const float* __restrict__ X, const float* __restrict__ W,
    const float* __restrict__ bias, const float* __restrict__ res,
    float* __restrict__ Y, int M, int Kc, int Nc, int act)
{
    __shared__ float As[16][36];                 // row stride 144B: B128-aligned
    const int tid  = threadIdx.x;
    const int wave = tid >> 5;
    const int lane = tid & 31;
    const int half = lane >> 4;
    const int m16  = lane & 15;
    const int mbase = blockIdx.y * 16;
    const int nbase = (blockIdx.x * 2 + wave) * 16;

    // staging ownership: thread -> row (tid>>2), 8 contiguous cols
    const int sr = tid >> 2;
    const int sc = (tid & 3) << 3;

    v8f acc;
    #pragma unroll
    for (int r = 0; r < 8; ++r) acc[r] = 0.0f;

    for (int kk = 0; kk < Kc; kk += 32) {
        const float* src = X + (size_t)(mbase + sr) * Kc + (kk + sc);
#if USE_ASYNC_LDS
        __builtin_amdgcn_global_load_async_to_lds_b128(
            (gv4i_p)(src), (lv4i_p)&As[sr][sc], 0, 0);
        __builtin_amdgcn_global_load_async_to_lds_b128(
            (gv4i_p)(src + 4), (lv4i_p)&As[sr][sc + 4], 0, 0);
        wait_async_lds();
#else
        float4 u0 = *(const float4*)(src);
        float4 u1 = *(const float4*)(src + 4);
        *(float4*)&As[sr][sc]     = u0;
        *(float4*)&As[sr][sc + 4] = u1;
#endif
        __syncthreads();

        // A fragment: lane = M row; per-lane K runs are contiguous:
        // j=0..7 -> K = half*8 + j ; j=8..15 -> K = 16 + half*8 + (j-8)
        const float* arow = &As[m16][half * 8];
        float4 a0 = *(const float4*)(arow);
        float4 a1 = *(const float4*)(arow + 4);
        float4 a2 = *(const float4*)(arow + 16);
        float4 a3 = *(const float4*)(arow + 20);
        v16h a;
        a[0]=(_Float16)a0.x; a[1]=(_Float16)a0.y; a[2]=(_Float16)a0.z; a[3]=(_Float16)a0.w;
        a[4]=(_Float16)a1.x; a[5]=(_Float16)a1.y; a[6]=(_Float16)a1.z; a[7]=(_Float16)a1.w;
        a[8]=(_Float16)a2.x; a[9]=(_Float16)a2.y; a[10]=(_Float16)a2.z; a[11]=(_Float16)a2.w;
        a[12]=(_Float16)a3.x; a[13]=(_Float16)a3.y; a[14]=(_Float16)a3.z; a[15]=(_Float16)a3.w;

        // B fragment: lane = N column of W (lane&15), elements = K rows in A's order
        const float* wp = W + (size_t)(kk + half * 8) * Nc + nbase + m16;
        v16h b;
        #pragma unroll
        for (int j = 0; j < 8; ++j) {
            b[j]     = (_Float16)wp[(size_t)j * Nc];
            b[j + 8] = (_Float16)wp[(size_t)(16 + j) * Nc];
        }

        if (kk + 32 < Kc)   // pull next W panel toward the cache
            __builtin_prefetch(W + (size_t)(kk + 32 + lane) * Nc + nbase);

        acc = __builtin_amdgcn_wmma_f32_16x16x32_f16(
                  false, a, false, b, (short)0, acc, false, false);
        __syncthreads();
    }

    // D layout: VGPR r -> row = half*8 + r, col = lane&15
    #pragma unroll
    for (int r = 0; r < 8; ++r) {
        int row = mbase + half * 8 + r;
        int col = nbase + m16;
        float v = acc[r];
        if (bias) v += bias[col];
        if (res)  v += res[(size_t)row * Nc + col];
        if (act)  v = v > 0.0f ? v : 0.0f;
        Y[(size_t)row * Nc + col] = v;
    }
}

// ============================ small naive GEMM =============================
__global__ __launch_bounds__(TPB) void k_small_gemm(
    const float* __restrict__ X, const float* __restrict__ W,
    const float* __restrict__ bias, float* __restrict__ Y,
    int M, int Kc, int Nc, int act)
{
    int t = blockIdx.x * TPB + threadIdx.x;
    if (t >= M * Nc) return;
    int j = t % Nc, i = t / Nc;
    float s = bias ? bias[j] : 0.0f;
    for (int k = 0; k < Kc; ++k) s += X[(size_t)i * Kc + k] * W[(size_t)k * Nc + j];
    if (act) s = s > 0.0f ? s : 0.0f;
    Y[t] = s;
}

// ============================ BatchNorm ====================================
__global__ __launch_bounds__(TPB) void k_bn_stats(
    const float* __restrict__ x, float* __restrict__ mean,
    float* __restrict__ rstd, int M, int C)
{
    __shared__ float ss[TPB], sq[TPB];
    int c = blockIdx.x, tid = threadIdx.x;
    float s = 0.0f, q = 0.0f;
    for (int i = tid; i < M; i += TPB) {
        float v = x[(size_t)i * C + c];
        s += v; q += v * v;
    }
    ss[tid] = s; sq[tid] = q; __syncthreads();
    for (int off = TPB / 2; off > 0; off >>= 1) {
        if (tid < off) { ss[tid] += ss[tid + off]; sq[tid] += sq[tid + off]; }
        __syncthreads();
    }
    if (tid == 0) {
        float m = ss[0] / (float)M;
        float v = sq[0] / (float)M - m * m;
        if (v < 0.0f) v = 0.0f;
        mean[c] = m;
        rstd[c] = rsqrtf(v + 1e-5f);
    }
}

__global__ __launch_bounds__(TPB) void k_bn_apply(
    float* __restrict__ x, const float* __restrict__ g,
    const float* __restrict__ bt, const float* __restrict__ mean,
    const float* __restrict__ rstd, int total, int C, int act)
{
    int t = blockIdx.x * TPB + threadIdx.x;
    if (t >= total) return;
    int c = t % C;
    float v = g[c] * (x[t] - mean[c]) * rstd[c] + bt[c];
    if (act) v = v > 0.0f ? v : 0.0f;
    x[t] = v;
}

// ============================ kNN (top-16 smallest sqdist) =================
__global__ __launch_bounds__(TPB) void k_knn(
    const float* __restrict__ qx, const float* __restrict__ sx,
    int* __restrict__ idx, int Nq, int Ns)
{
    int t = blockIdx.x * TPB + threadIdx.x;
    if (t >= 32 * Nq) return;
    int b = t / Nq;
    float px = qx[(size_t)t * 3 + 0];
    float py = qx[(size_t)t * 3 + 1];
    float pz = qx[(size_t)t * 3 + 2];
    float bd[16]; int bi[16];
    #pragma unroll
    for (int j = 0; j < 16; ++j) { bd[j] = 3.4e38f; bi[j] = 0; }
    const float* sp = sx + (size_t)b * Ns * 3;
    for (int m = 0; m < Ns; ++m) {
        float dx = px - sp[m * 3 + 0];
        float dy = py - sp[m * 3 + 1];
        float dz = pz - sp[m * 3 + 2];
        float cd = dx * dx + dy * dy + dz * dz;
        int ci = m;
        #pragma unroll
        for (int j = 0; j < 16; ++j) {      // register-resident insertion chain
            if (cd < bd[j]) {
                float tf = bd[j]; bd[j] = cd; cd = tf;
                int   ti = bi[j]; bi[j] = ci; ci = ti;
            }
        }
    }
    #pragma unroll
    for (int j = 0; j < 16; ++j) idx[(size_t)t * 16 + j] = bi[j];
}

// ============================ Farthest point sampling ======================
__global__ __launch_bounds__(TPB) void k_fps(
    const float* __restrict__ xyz, int* __restrict__ fidx, int N, int np)
{
    __shared__ float dist[1024];
    __shared__ float rv[TPB];
    __shared__ int   ri[TPB];
    __shared__ int   scur;
    int b = blockIdx.x, tid = threadIdx.x;
    const float* xb = xyz + (size_t)b * N * 3;
    for (int i = tid; i < N; i += TPB) dist[i] = 1e10f;
    if (tid == 0) { fidx[b * np] = 0; scur = 0; }
    __syncthreads();
    for (int s = 1; s < np; ++s) {
        int cur = scur;
        float cx = xb[cur * 3], cy = xb[cur * 3 + 1], cz = xb[cur * 3 + 2];
        float bv = -1.0f; int bi = 0;
        for (int i = tid; i < N; i += TPB) {
            float dx = xb[i * 3] - cx, dy = xb[i * 3 + 1] - cy, dz = xb[i * 3 + 2] - cz;
            float nd = fminf(dist[i], dx * dx + dy * dy + dz * dz);
            dist[i] = nd;
            if (nd > bv) { bv = nd; bi = i; }   // strict > keeps first index
        }
        rv[tid] = bv; ri[tid] = bi;
        __syncthreads();
        for (int off = TPB / 2; off > 0; off >>= 1) {
            if (tid < off) {
                float ov = rv[tid + off]; int oi = ri[tid + off];
                if (ov > rv[tid] || (ov == rv[tid] && oi < ri[tid])) {
                    rv[tid] = ov; ri[tid] = oi;
                }
            }
            __syncthreads();
        }
        if (tid == 0) { fidx[b * np + s] = ri[0]; scur = ri[0]; }
        __syncthreads();
    }
}

// ============================ misc elementwise kernels =====================
// hidden = relu( (xyz[n]-xyz[nbr]) @ p1w + p1b ), 64 outputs per (b,n,k) row
__global__ __launch_bounds__(TPB) void k_pos_hidden(
    const float* __restrict__ xyz, const int* __restrict__ idx,
    const float* __restrict__ p1w, const float* __restrict__ p1b,
    float* __restrict__ hid, int row0, int rows, int NL)
{
    int t = blockIdx.x * TPB + threadIdx.x;
    if (t >= rows * 64) return;
    int r = t >> 6, j = t & 63;
    int rg = row0 + r;
    int b  = rg / (NL * 16);
    int n  = (rg / 16) % NL;
    int nb = idx[rg];
    const float* pa = xyz + (size_t)(b * NL + n)  * 3;
    const float* pb = xyz + (size_t)(b * NL + nb) * 3;
    float rx = pa[0] - pb[0], ry = pa[1] - pb[1], rz = pa[2] - pb[2];
    float v = rx * p1w[j] + ry * p1w[64 + j] + rz * p1w[128 + j] + p1b[j];
    hid[(size_t)r * 64 + j] = v > 0.0f ? v : 0.0f;
}

// t = q[b,n] - k[b,idx] + pos
__global__ __launch_bounds__(TPB) void k_build_t(
    const float* __restrict__ q, const float* __restrict__ kf,
    const int* __restrict__ idx, const float* __restrict__ pos,
    float* __restrict__ tt, int NL, int d, int total)
{
    int t = blockIdx.x * TPB + threadIdx.x;
    if (t >= total) return;
    int c  = t % d;
    int rg = t / d;
    int b  = rg / (NL * 16);
    int n  = (rg / 16) % NL;
    int nb = idx[rg];
    tt[t] = q[(size_t)(b * NL + n) * d + c] - kf[(size_t)(b * NL + nb) * d + c] + pos[t];
}

// softmax over the K=16 axis, with pre-scale 1/sqrt(d)
__global__ __launch_bounds__(TPB) void k_softmax_k(
    float* __restrict__ a, int d, float scale, int total)
{
    int t = blockIdx.x * TPB + threadIdx.x;
    if (t >= total) return;
    int c  = t % d;
    int bn = t / d;
    size_t base = (size_t)bn * 16 * d + c;
    float v[16], mx = -3.4e38f;
    #pragma unroll
    for (int k = 0; k < 16; ++k) { v[k] = a[base + (size_t)k * d] * scale; mx = fmaxf(mx, v[k]); }
    float s = 0.0f;
    #pragma unroll
    for (int k = 0; k < 16; ++k) { v[k] = expf(v[k] - mx); s += v[k]; }
    float inv = 1.0f / s;
    #pragma unroll
    for (int k = 0; k < 16; ++k) a[base + (size_t)k * d] = v[k] * inv;
}

// new = sum_k attn * (v[b,idx] + pos)
__global__ __launch_bounds__(TPB) void k_aggregate(
    const float* __restrict__ a, const float* __restrict__ vf,
    const int* __restrict__ idx, const float* __restrict__ pos,
    float* __restrict__ out, int NL, int d, int total)
{
    int t = blockIdx.x * TPB + threadIdx.x;
    if (t >= total) return;
    int c  = t % d;
    int bn = t / d;
    int b  = bn / NL;
    size_t base = (size_t)bn * 16 * d + c;
    float s = 0.0f;
    #pragma unroll
    for (int k = 0; k < 16; ++k) {
        int nb = idx[(size_t)bn * 16 + k];
        s += a[base + (size_t)k * d] *
             (vf[(size_t)(b * NL + nb) * d + c] + pos[base + (size_t)k * d]);
    }
    out[t] = s;
}

__global__ __launch_bounds__(TPB) void k_gather_rows(
    float* __restrict__ out, const float* __restrict__ src,
    const int* __restrict__ idx, int rowsPerBatch, int Nsrc, int C, int totalRows)
{
    int t = blockIdx.x * TPB + threadIdx.x;
    if (t >= totalRows * C) return;
    int c = t % C, i = t / C;
    int b = i / rowsPerBatch;
    out[t] = src[(size_t)(b * Nsrc + idx[i]) * C + c];
}

__global__ __launch_bounds__(TPB) void k_maxpool_k(
    const float* __restrict__ in, float* __restrict__ out, int C, int total)
{
    int t = blockIdx.x * TPB + threadIdx.x;
    if (t >= total) return;
    int c = t % C, bn = t / C;
    float m = -3.4e38f;
    #pragma unroll
    for (int k = 0; k < 16; ++k) m = fmaxf(m, in[((size_t)bn * 16 + k) * C + c]);
    out[t] = m;
}

__global__ __launch_bounds__(TPB) void k_rowmean(
    const float* __restrict__ in, float* __restrict__ out, int Npts, int C, int total)
{
    int t = blockIdx.x * TPB + threadIdx.x;
    if (t >= total) return;
    int c = t % C, b = t / C;
    float s = 0.0f;
    for (int n = 0; n < Npts; ++n) s += in[((size_t)b * Npts + n) * C + c];
    out[t] = s / (float)Npts;
}

// ============================ host orchestration ===========================
static inline dim3 g1(long long n) { return dim3((unsigned)((n + TPB - 1) / TPB)); }

extern "C" void kernel_launch(void* const* d_in, const int* in_sizes, int n_in,
                              void* d_out, int out_size, void* d_ws, size_t ws_size,
                              hipStream_t stream)
{
    (void)in_sizes; (void)n_in; (void)out_size; (void)ws_size;

    const float* xyz_in = (const float*)d_in[0];
    const float* x_in   = (const float*)d_in[1];
    auto P = [&](int i) { return (const float*)d_in[i]; };
    // input order (insertion order of setup_inputs dict):
    // 0 xyz, 1 x, 2..5 fc1{w,b,g,bt},
    // 6..18 enc1{wq,wk,wv,p1w,p1b,p2w,p2b,a1w,a1b,a2w,a2b,lfw,lfb}, 19..22 td1{w,b,g,bt},
    // 23..35 enc2, 36..39 td2, 40..52 enc3, 53..56 td3, 57..69 enc4,
    // 70..79 f2w1,f2b1,f2g1,f2t1,f2w2,f2b2,f2g2,f2t2,f2w3,f2b3

    // ---- workspace layout (floats) --------------------------------------
    float* ws = (float*)d_ws;
    size_t off = 0;
    auto alloc = [&](size_t n) { float* p = ws + off; off += n; return p; };
    float* featA = alloc(1u << 20);       // current features [B*NL, d]
    float* xyzA  = alloc(32 * 1024 * 3);
    float* xyzB  = alloc(32 * 1024 * 3);
    float* qb    = alloc(1u << 20);
    float* kb    = alloc(1u << 20);
    float* vb    = alloc(1u << 20);
    float* posb  = alloc(1u << 24);       // [BNK, d]  (64 MB)
    float* tb    = alloc(1u << 24);       // [BNK, d]  (64 MB)
    float* attn  = alloc(1u << 24);       // [BNK, d]  (64 MB)
    float* hid   = alloc(1u << 24);       // chunked hidden (64 MB)
    float* newb  = alloc(1u << 20);
    float* meanb = alloc(1024);
    float* rstdb = alloc(1024);
    int*   idxb  = (int*)alloc(524288);   // kNN indices [B*NL*K]
    int*   fidxb = (int*)alloc(32 * 512); // FPS indices

    auto gemmW = [&](const float* X, const float* W, const float* bias,
                     const float* res, float* Y, int M, int Kc, int Nc, int act) {
        dim3 g((unsigned)(Nc / 32), (unsigned)(M / 16));
        k_wmma_gemm<<<g, dim3(64), 0, stream>>>(X, W, bias, res, Y, M, Kc, Nc, act);
    };

    // ---- fc1 + BN + relu -------------------------------------------------
    k_small_gemm<<<g1(32768LL * 32), TPB, 0, stream>>>(x_in, P(2), P(3), featA, 32768, 6, 32, 0);
    k_bn_stats<<<32, TPB, 0, stream>>>(featA, meanb, rstdb, 32768, 32);
    k_bn_apply<<<g1(32768LL * 32), TPB, 0, stream>>>(featA, P(4), P(5), meanb, rstdb, 32768 * 32, 32, 1);

    // ---- point-transformer layer ----------------------------------------
    auto pt_layer = [&](const float* xyz, float* feat, int NL, int d, int pb) {
        int BN = 32 * NL, BNK = BN * 16;
        k_knn<<<g1(BN), TPB, 0, stream>>>(xyz, xyz, idxb, NL, NL);
        gemmW(feat, P(pb + 0), nullptr, nullptr, qb, BN, d, d, 0);   // q
        gemmW(feat, P(pb + 1), nullptr, nullptr, kb, BN, d, d, 0);   // k
        gemmW(feat, P(pb + 2), nullptr, nullptr, vb, BN, d, d, 0);   // v
        // position encoding MLP: relu(rel@p1w+b) @ p2w + b
        const int pchunk = 262144;                // rows; pchunk*64 fits hid
        for (int r0 = 0; r0 < BNK; r0 += pchunk) {
            int rows = BNK - r0 < pchunk ? BNK - r0 : pchunk;
            k_pos_hidden<<<g1((long long)rows * 64), TPB, 0, stream>>>(
                xyz, idxb, P(pb + 3), P(pb + 4), hid, r0, rows, NL);
            gemmW(hid, P(pb + 5), P(pb + 6), nullptr, posb + (size_t)r0 * d, rows, 64, d, 0);
        }
        // attention scores: a = relu((q-k+pos)@a1w+b) @ a2w + b
        k_build_t<<<g1((long long)BNK * d), TPB, 0, stream>>>(qb, kb, idxb, posb, tb, NL, d, BNK * d);
        int H = 4 * d;
        int achunk = (1 << 24) / H;               // rows per chunk (multiple of 16)
        for (int r0 = 0; r0 < BNK; r0 += achunk) {
            int rows = BNK - r0 < achunk ? BNK - r0 : achunk;
            gemmW(tb + (size_t)r0 * d, P(pb + 7), P(pb + 8), nullptr, hid, rows, d, H, 1);
            gemmW(hid, P(pb + 9), P(pb + 10), nullptr, attn + (size_t)r0 * d, rows, H, d, 0);
        }
        float scale = 1.0f / sqrtf((float)d);
        k_softmax_k<<<g1((long long)BN * d), TPB, 0, stream>>>(attn, d, scale, BN * d);
        k_aggregate<<<g1((long long)BN * d), TPB, 0, stream>>>(attn, vb, idxb, posb, newb, NL, d, BN * d);
        gemmW(newb, P(pb + 11), P(pb + 12), feat, feat, BN, d, d, 0); // + residual
    };

    // ---- transition down -------------------------------------------------
    auto tdown = [&](const float* xyz, float* nxyz, float* feat,
                     int NL, int Nn, int Ci, int Co, int pb) {
        k_fps<<<32, TPB, 0, stream>>>(xyz, fidxb, NL, Nn);
        k_gather_rows<<<g1(32LL * Nn * 3), TPB, 0, stream>>>(nxyz, xyz, fidxb, Nn, NL, 3, 32 * Nn);
        k_knn<<<g1(32LL * Nn), TPB, 0, stream>>>(nxyz, xyz, idxb, Nn, NL);
        int Mrows = 32 * Nn * 16;
        k_gather_rows<<<g1((long long)Mrows * Ci), TPB, 0, stream>>>(tb, feat, idxb, Nn * 16, NL, Ci, Mrows);
        gemmW(tb, P(pb + 0), P(pb + 1), nullptr, attn, Mrows, Ci, Co, 0);
        k_bn_stats<<<Co, TPB, 0, stream>>>(attn, meanb, rstdb, Mrows, Co);
        k_bn_apply<<<g1((long long)Mrows * Co), TPB, 0, stream>>>(attn, P(pb + 2), P(pb + 3), meanb, rstdb, Mrows * Co, Co, 1);
        k_maxpool_k<<<g1(32LL * Nn * Co), TPB, 0, stream>>>(attn, feat, Co, 32 * Nn * Co);
    };

    // ---- network ---------------------------------------------------------
    pt_layer(xyz_in, featA, 1024, 32, 6);
    tdown(xyz_in, xyzA, featA, 1024, 512, 32, 64, 19);
    pt_layer(xyzA, featA, 512, 64, 23);
    tdown(xyzA, xyzB, featA, 512, 128, 64, 128, 36);
    pt_layer(xyzB, featA, 128, 128, 40);
    tdown(xyzB, xyzA, featA, 128, 32, 128, 256, 53);
    pt_layer(xyzA, featA, 32, 256, 57);

    // ---- classification head --------------------------------------------
    k_rowmean<<<g1(32LL * 256), TPB, 0, stream>>>(featA, newb, 32, 256, 32 * 256);
    k_small_gemm<<<g1(32LL * 256), TPB, 0, stream>>>(newb, P(70), P(71), qb, 32, 256, 256, 0);
    k_bn_stats<<<256, TPB, 0, stream>>>(qb, meanb, rstdb, 32, 256);
    k_bn_apply<<<g1(32LL * 256), TPB, 0, stream>>>(qb, P(72), P(73), meanb, rstdb, 32 * 256, 256, 1);
    k_small_gemm<<<g1(32LL * 128), TPB, 0, stream>>>(qb, P(74), P(75), kb, 32, 256, 128, 0);
    k_bn_stats<<<128, TPB, 0, stream>>>(kb, meanb, rstdb, 32, 128);
    k_bn_apply<<<g1(32LL * 128), TPB, 0, stream>>>(kb, P(76), P(77), meanb, rstdb, 32 * 128, 128, 1);
    k_small_gemm<<<g1(32LL * 40), TPB, 0, stream>>>(kb, P(78), P(79), (float*)d_out, 32, 128, 40, 0);
}
